// Attention_85323820302479
// MI455X (gfx1250) — compile-verified, matrix operand order checked by reading, not run on hardware
//
#include <hip/hip_runtime.h>

typedef _Float16 f16;
typedef _Float16 v16h __attribute__((ext_vector_type(16)));
typedef _Float16 v8h  __attribute__((ext_vector_type(8)));
typedef float    v8f  __attribute__((ext_vector_type(8)));
typedef int      v4i  __attribute__((ext_vector_type(4)));

typedef __attribute__((address_space(1))) v4i* gv4i_ptr;  // global
typedef __attribute__((address_space(3))) v4i* lv4i_ptr;  // LDS

#define NUM_H   16
#define HDIM    128
#define NTOK    2048
#define HID     2048
#define BATCH   2
#define M_TOT   (BATCH * NTOK)        // 4096
#define QKV_N   (3 * NUM_H * HDIM)    // 6144

// ---------------------------------------------------------------------------
// CDNA5 async global->LDS copy (ASYNCcnt-tracked), with fallback.
// Probe-discovered signature: (v4i addrspace(1)*, v4i addrspace(3)*, imm, imm)
// ---------------------------------------------------------------------------
#if __has_builtin(__builtin_amdgcn_global_load_async_to_lds_b128)
#define USE_ASYNC 1
#else
#define USE_ASYNC 0
#endif

__device__ __forceinline__ void cp16(f16* lds_dst, const f16* gsrc) {
#if USE_ASYNC
    __builtin_amdgcn_global_load_async_to_lds_b128(
        (gv4i_ptr)gsrc, (lv4i_ptr)lds_dst, 0, 0);
#else
    *(v8h*)lds_dst = *(const v8h*)gsrc;
#endif
}

__device__ __forceinline__ void wait_async_stage() {
#if USE_ASYNC
#if __has_builtin(__builtin_amdgcn_s_wait_asynccnt)
    __builtin_amdgcn_s_wait_asynccnt(4);
#else
    asm volatile("s_wait_asynccnt 4" ::: "memory");
#endif
#endif
}

__device__ __forceinline__ void wait_async_all() {
#if USE_ASYNC
#if __has_builtin(__builtin_amdgcn_s_wait_asynccnt)
    __builtin_amdgcn_s_wait_asynccnt(0);
#else
    asm volatile("s_wait_asynccnt 0" ::: "memory");
#endif
#endif
}

// ---------------------------------------------------------------------------
// Fragment helpers. CDNA5 WMMA f16 A-fragment (16x32, M on lanes):
//   lane L: row = L%16, half = L/16
//   elements 0..7  : k = k0 + half*8 + j        (contiguous 16 bytes)
//   elements 8..15 : k = k0 + 16 + half*8 + j   (contiguous 16 bytes)
// B-fragment mirrors with N on lanes, so K-contiguous [n][k] storage uses the
// same loader.
// ---------------------------------------------------------------------------
__device__ __forceinline__ v16h make_frag(const f16* p) {
    v8h lo = *(const v8h*)(p);
    v8h hi = *(const v8h*)(p + 16);
    return __builtin_shufflevector(lo, hi, 0, 1, 2, 3, 4, 5, 6, 7,
                                           8, 9, 10, 11, 12, 13, 14, 15);
}

__device__ __forceinline__ v16h load_frag(const f16* base, int ld, int row,
                                          int k0, int lane) {
    const f16* p = base + (size_t)(row + (lane & 15)) * ld + k0 + ((lane >> 4) << 3);
    return make_frag(p);
}

__device__ __forceinline__ v8f wmma16(v16h a, v16h b, v8f c) {
    return __builtin_amdgcn_wmma_f32_16x16x32_f16(false, a, false, b,
                                                  (short)0, c, false, false);
}

// ---------------------------------------------------------------------------
// Shared double-buffered 128x128 GEMM core (K = HID, K-step 32).
// Block: 256 threads = 8 waves arranged 2(M) x 4(N); each wave owns a 64x32
// output tile = 4x2 WMMA tiles. Async copies for stage k+1 are issued before
// computing on stage k, then s_wait_asynccnt 4 retires stage k only.
// ---------------------------------------------------------------------------
__device__ __forceinline__ void gemm128x128(
    const f16* __restrict__ Amat, const f16* __restrict__ Bmat,
    int tileM, int tileN,
    f16 (*As)[128][32], f16 (*Bs)[128][32],
    int tid, int lane, v8f acc[4][2]) {
    const int wave = tid >> 5;
    const int wm2 = wave >> 2;   // 0..1
    const int wn4 = wave & 3;    // 0..3

    // stage(kt, buf): 128x32 A chunks + 128x32 B chunks, 4 b128 copies/thread
#define STAGE(kt, buf)                                                         \
    {                                                                          \
        _Pragma("unroll")                                                      \
        for (int i_ = 0; i_ < 2; ++i_) {                                       \
            int idx_ = tid + i_ * 256;                                         \
            int r_ = idx_ >> 2, cc_ = (idx_ & 3) << 3;                         \
            cp16(&As[buf][r_][cc_],                                            \
                 Amat + (size_t)(tileM + r_) * HID + (kt) + cc_);              \
            cp16(&Bs[buf][r_][cc_],                                            \
                 Bmat + (size_t)(tileN + r_) * HID + (kt) + cc_);              \
        }                                                                      \
    }

    STAGE(0, 0);
    for (int k = 0; k < HID / 32; ++k) {
        const int cur = k & 1;
        if (k + 1 < HID / 32) {
            STAGE((k + 1) * 32, cur ^ 1);
            wait_async_stage();   // retire stage k (in-order), leave k+1 in flight
        } else {
            wait_async_all();
        }
        __syncthreads();
        v16h a[4], b[2];
#pragma unroll
        for (int i = 0; i < 4; ++i)
            a[i] = load_frag(&As[cur][0][0], 32, wm2 * 64 + i * 16, 0, lane);
#pragma unroll
        for (int j = 0; j < 2; ++j)
            b[j] = load_frag(&Bs[cur][0][0], 32, wn4 * 32 + j * 16, 0, lane);
#pragma unroll
        for (int i = 0; i < 4; ++i)
#pragma unroll
            for (int j = 0; j < 2; ++j)
                acc[i][j] = wmma16(a[i], b[j], acc[i][j]);
        __syncthreads();   // all waves done reading buf[cur] before overwrite
    }
#undef STAGE
}

// ---------------------------------------------------------------------------
// Prep kernels
// ---------------------------------------------------------------------------
__global__ __launch_bounds__(256) void cvt_f32_f16_kernel(
    const float* __restrict__ src, f16* __restrict__ dst, int n) {
    int i = blockIdx.x * 256 + threadIdx.x;
    if (i < n) dst[i] = (f16)src[i];
}

// src: R x C f32 row-major  ->  dst: C x R f16 row-major (transpose + cast)
__global__ __launch_bounds__(256) void transpose_f32_f16_kernel(
    const float* __restrict__ src, f16* __restrict__ dst, int R, int C) {
    int i = blockIdx.x * 256 + threadIdx.x;
    if (i < R * C) {
        int c = i / R;
        int r = i - c * R;
        dst[(size_t)c * R + r] = (f16)src[(size_t)r * C + c];
    }
}

// ---------------------------------------------------------------------------
// Kernel 1: qkv = x @ qkv_w, fused RoPE, scatter to q/k (n-major), v (d-major)
// ---------------------------------------------------------------------------
__global__ __launch_bounds__(256) void qkv_rope_kernel(
    const f16* __restrict__ xh,       // [4096][2048]
    const f16* __restrict__ wT,       // [6144][2048]  (qkv_w transposed)
    const float* __restrict__ rope,   // [2048][128]
    f16* __restrict__ qb,             // [B][H][N][D]
    f16* __restrict__ kb,             // [B][H][N][D]
    f16* __restrict__ vT) {           // [B][H][D][N]
    __shared__ f16 As[2][128][32];
    __shared__ f16 Bs[2][128][32];

    const int tid  = threadIdx.x;
    const int lane = tid & 31;
    const int wave = tid >> 5;
    const int halfl = lane >> 4;
    const int tileM = blockIdx.y * 128;
    const int tileN = blockIdx.x * 128;
    const int wm2 = wave >> 2;
    const int wn4 = wave & 3;

    v8f acc[4][2] = {};
    gemm128x128(xh, wT, tileM, tileN, As, Bs, tid, lane, acc);

    // Epilogue: RoPE on q/k, scatter. Columns of one WMMA tile share (s,h),
    // so the q/k/v branch is wave-uniform within each (i,j).
#pragma unroll
    for (int i = 0; i < 4; ++i) {
#pragma unroll
        for (int j = 0; j < 2; ++j) {
            int gc = tileN + wn4 * 32 + j * 16 + (lane & 15);
            int s = gc >> 11;            // 0=q 1=k 2=v
            int h = (gc >> 7) & 15;
            int d = gc & 127;
            int rowbase = tileM + wm2 * 64 + i * 16 + halfl * 8;
#pragma unroll
            for (int r = 0; r < 8; ++r) {
                int gm = rowbase + r;
                int b  = gm >> 11;
                int n  = gm & 2047;
                float val = acc[i][j][r];
                if (s == 2) {
                    vT[(((size_t)(b * NUM_H + h) * HDIM + d) * NTOK) + n] = (f16)val;
                } else {
                    float cs = rope[n * HDIM + (d & ~1)];
                    float sn = rope[n * HDIM + (d & ~1) + 1];
                    float partner = __shfl_xor(val, 1, 32);
                    float o = (d & 1) ? (val * cs + partner * sn)
                                      : (val * cs - partner * sn);
                    f16* dst = (s == 0) ? qb : kb;
                    dst[((size_t)(b * NUM_H + h) * NTOK + n) * HDIM + d] = (f16)o;
                }
            }
        }
    }
}

// ---------------------------------------------------------------------------
// Kernel 2: flash attention. One wave per 16-row q tile; 8 waves / block.
// Keys streamed in blocks of 32: 8 QK^T WMMAs + online softmax + 8 P*V WMMAs.
// The P re-layout LDS tile is wave-private: same-wave DS ops are in-order and
// the compiler inserts s_wait_dscnt for the aliasing load, so no barriers.
// ---------------------------------------------------------------------------
__global__ __launch_bounds__(256) void attn_kernel(
    const f16* __restrict__ qb,   // [B][H][N][D]
    const f16* __restrict__ kb,   // [B][H][N][D]
    const f16* __restrict__ vT,   // [B][H][D][N]
    f16* __restrict__ ao) {       // [B][N][H*D]
    __shared__ f16 Pl[8][16][32]; // per-wave P tile (re-layout buffer)

    const int tid  = threadIdx.x;
    const int lane = tid & 31;
    const int wave = tid >> 5;
    const int halfl = lane >> 4;
    const int b = blockIdx.z;
    const int h = blockIdx.y;
    const int q0 = blockIdx.x * 128 + wave * 16;

    const f16* Q = qb + (size_t)(b * NUM_H + h) * NTOK * HDIM;
    const f16* K = kb + (size_t)(b * NUM_H + h) * NTOK * HDIM;
    const f16* V = vT + (size_t)(b * NUM_H + h) * HDIM * NTOK;

    // Q fragments for the whole D=128 (4 chunks of K=32), kept resident.
    v16h qa[4];
#pragma unroll
    for (int kd = 0; kd < 4; ++kd) qa[kd] = load_frag(Q, HDIM, q0, kd * 32, lane);

    v8f o[8] = {};
    float mi[8], li[8];
#pragma unroll
    for (int r = 0; r < 8; ++r) { mi[r] = -1e30f; li[r] = 0.0f; }
    const float sc = 0.088388347648318447f; // 1/sqrt(128)

    for (int kblk = 0; kblk < NTOK; kblk += 32) {
        if (kblk + 32 < NTOK) {   // hint next K/V block into cache
            __builtin_prefetch(K + (size_t)(kblk + 32) * HDIM + (lane << 2), 0, 0);
            __builtin_prefetch(V + (size_t)(lane << 3) * NTOK + kblk + 32, 0, 0);
        }
        v8f s0 = {}, s1 = {};
#pragma unroll
        for (int kd = 0; kd < 4; ++kd) {
            v16h b0 = load_frag(K, HDIM, kblk,      kd * 32, lane);
            v16h b1 = load_frag(K, HDIM, kblk + 16, kd * 32, lane);
            s0 = wmma16(qa[kd], b0, s0);
            s1 = wmma16(qa[kd], b1, s1);
        }
        // Online softmax. Row r lives in element r of each accumulator for the
        // 16 lanes of this half-wave; reduce over cols with intra-half shfl_xor.
#pragma unroll
        for (int r = 0; r < 8; ++r) {
            float a0 = s0[r] * sc;
            float a1 = s1[r] * sc;
            float m = fmaxf(a0, a1);
#pragma unroll
            for (int off = 1; off < 16; off <<= 1)
                m = fmaxf(m, __shfl_xor(m, off, 32));
            float mnew = fmaxf(mi[r], m);
            float alpha = __expf(mi[r] - mnew);
            mi[r] = mnew;
            float p0 = __expf(a0 - mnew);
            float p1 = __expf(a1 - mnew);
            float rs = p0 + p1;
#pragma unroll
            for (int off = 1; off < 16; off <<= 1)
                rs += __shfl_xor(rs, off, 32);
            li[r] = li[r] * alpha + rs;
#pragma unroll
            for (int ct = 0; ct < 8; ++ct) o[ct][r] *= alpha;
            Pl[wave][halfl * 8 + r][lane & 15]        = (f16)p0;
            Pl[wave][halfl * 8 + r][16 + (lane & 15)] = (f16)p1;
        }
        v16h pa = load_frag(&Pl[wave][0][0], 32, 0, 0, lane);
#pragma unroll
        for (int ct = 0; ct < 8; ++ct) {
            v16h bv = load_frag(V, NTOK, ct * 16, kblk, lane); // V^T rows = d
            o[ct] = wmma16(pa, bv, o[ct]);
        }
    }

#pragma unroll
    for (int ct = 0; ct < 8; ++ct) {
        int d = ct * 16 + (lane & 15);
#pragma unroll
        for (int r = 0; r < 8; ++r) {
            int n = q0 + halfl * 8 + r;
            ao[((size_t)(b * NTOK + n)) * HID + h * HDIM + d] =
                (f16)(o[ct][r] * (1.0f / li[r]));
        }
    }
}

// ---------------------------------------------------------------------------
// Kernel 3: out = attn_out @ out_w  (f32 result).
// ---------------------------------------------------------------------------
__global__ __launch_bounds__(256) void out_proj_kernel(
    const f16* __restrict__ ao,    // [4096][2048]
    const f16* __restrict__ wT,    // [2048][2048] (out_w transposed)
    float* __restrict__ out) {     // [4096][2048]
    __shared__ f16 As[2][128][32];
    __shared__ f16 Bs[2][128][32];

    const int tid  = threadIdx.x;
    const int lane = tid & 31;
    const int wave = tid >> 5;
    const int halfl = lane >> 4;
    const int tileM = blockIdx.y * 128;
    const int tileN = blockIdx.x * 128;
    const int wm2 = wave >> 2;
    const int wn4 = wave & 3;

    v8f acc[4][2] = {};
    gemm128x128(ao, wT, tileM, tileN, As, Bs, tid, lane, acc);

#pragma unroll
    for (int i = 0; i < 4; ++i) {
#pragma unroll
        for (int j = 0; j < 2; ++j) {
            int gc = tileN + wn4 * 32 + j * 16 + (lane & 15);
            int rowbase = tileM + wm2 * 64 + i * 16 + halfl * 8;
#pragma unroll
            for (int r = 0; r < 8; ++r) {
                out[(size_t)(rowbase + r) * HID + gc] = acc[i][j][r];
            }
        }
    }
}

// ---------------------------------------------------------------------------
// Launch
// ---------------------------------------------------------------------------
extern "C" void kernel_launch(void* const* d_in, const int* in_sizes, int n_in,
                              void* d_out, int out_size, void* d_ws, size_t ws_size,
                              hipStream_t stream) {
    const float* x      = (const float*)d_in[0];   // [2,2048,2048]
    const float* rope   = (const float*)d_in[1];   // [2048,1,128]
    const float* qkv_w  = (const float*)d_in[2];   // [2048,6144]
    const float* out_w  = (const float*)d_in[3];   // [2048,2048]
    float* out = (float*)d_out;

    char* ws = (char*)d_ws;
    f16* xh     = (f16*)(ws);                       // 16 MB
    f16* qkv_wT = (f16*)(ws + 16777216ull);         // 24 MB
    f16* out_wT = (f16*)(ws + 41943040ull);         //  8 MB
    f16* qb     = (f16*)(ws + 50331648ull);         // 16 MB
    f16* kbuf   = (f16*)(ws + 67108864ull);         // 16 MB
    f16* vT     = (f16*)(ws + 83886080ull);         // 16 MB
    f16* ao     = (f16*)(ws + 100663296ull);        // 16 MB

    cvt_f32_f16_kernel<<<(M_TOT * HID) / 256, 256, 0, stream>>>(x, xh, M_TOT * HID);
    transpose_f32_f16_kernel<<<(HID * QKV_N) / 256, 256, 0, stream>>>(qkv_w, qkv_wT, HID, QKV_N);
    transpose_f32_f16_kernel<<<(HID * HID) / 256, 256, 0, stream>>>(out_w, out_wT, HID, HID);

    qkv_rope_kernel<<<dim3(QKV_N / 128, M_TOT / 128), 256, 0, stream>>>(
        xh, qkv_wT, rope, qb, kbuf, vT);

    attn_kernel<<<dim3(NTOK / 128, NUM_H, BATCH), 256, 0, stream>>>(
        qb, kbuf, vT, ao);

    out_proj_kernel<<<dim3(HID / 128, M_TOT / 128), 256, 0, stream>>>(
        ao, out_wT, out);
}
